// ST_TGN_76055280877608
// MI455X (gfx1250) — compile-verified, matrix operand order checked by reading, not run on hardware
//
#include <hip/hip_runtime.h>

typedef __attribute__((ext_vector_type(16))) _Float16 v16h;
typedef __attribute__((ext_vector_type(8)))  _Float16 v8h;
typedef __attribute__((ext_vector_type(8)))  float    v8f;

#define NB_    8
#define WIN_   12
#define S_     1000
#define F_     16
#define HID_   128
#define NHEADS_ 4
#define HD_    32
#define DFF_   2048
#define NN_    (NB_*WIN_*S_)   /* 96000 nodes */
#define BT_    (NB_*WIN_)      /* 96 tokens   */
#define LN_EPS 1e-5f

// ---------------------------------------------------------------- utilities
__global__ void fill_zero_kernel(float* __restrict__ p, long n) {
    long i = (long)blockIdx.x * blockDim.x + threadIdx.x;
    if (i < n) p[i] = 0.f;
}

// count in-degree (dst occurrences incl. self loops) with fp32 atomics
__global__ void deg_count_kernel(const int* __restrict__ ei, float* __restrict__ deg, int E) {
    int e = blockIdx.x * blockDim.x + threadIdx.x;
    if (e < E)            atomicAdd(&deg[ei[E + e]], 1.0f);
    else if (e < E + NN_) atomicAdd(&deg[e - E],     1.0f);
}

__global__ void deg_to_dinv_kernel(float* __restrict__ deg) {
    int i = blockIdx.x * blockDim.x + threadIdx.x;
    if (i < NN_) { float d = deg[i]; deg[i] = (d > 0.f) ? rsqrtf(d) : 0.f; }
}

// pad x [N,16] -> xpad [N,32] with zeros in cols 16..31
__global__ void pad_x_kernel(const float* __restrict__ x, float* __restrict__ xpad) {
    long i = (long)blockIdx.x * blockDim.x + threadIdx.x;
    if (i >= (long)NN_ * 32) return;
    int row = (int)(i >> 5), c = (int)(i & 31);
    xpad[i] = (c < F_) ? x[(long)row * F_ + c] : 0.f;
}

// stage weight: B fp32 [Ksrc,Nc] -> Bt f16 [Nc,K] (transposed, zero-padded K>=Ksrc)
__global__ void conv_bt_kernel(const float* __restrict__ B, _Float16* __restrict__ Bt,
                               int K, int Nc, int Ksrc) {
    long i = (long)blockIdx.x * blockDim.x + threadIdx.x;
    if (i >= (long)Nc * K) return;
    int n = (int)(i / K), kk = (int)(i % K);
    Bt[i] = (kk < Ksrc) ? (_Float16)B[(long)kk * Nc + n] : (_Float16)0.f;
}

// ---------------------------------------------------------------- WMMA GEMM
// C[M,Nc] = A[M,K](fp32) @ Bt[Nc,K](f16, pre-transposed) (+ bias) (+ relu)
// grid = (M/16, Nc/64), block = 128 threads (4 waves, one 16x16 tile each).
template<int K>
__global__ __launch_bounds__(128) void wmma_gemm_kernel(
        const float* __restrict__ A, const _Float16* __restrict__ Bt,
        const float* __restrict__ bias, float* __restrict__ C,
        int Nc, int relu)
{
    const int lane = threadIdx.x & 31;
    const int wave = threadIdx.x >> 5;
    const int row0 = blockIdx.x * 16;
    const int col0 = (blockIdx.y * 4 + wave) * 16;
    const int lr   = lane & 15;            // A row / B col owned by this lane
    const int hk   = (lane >> 4) << 4;     // K sub-chunk: 0 or 16

    const float*    __restrict__ arow = A  + (long)(row0 + lr) * K + hk;
    const _Float16* __restrict__ brow = Bt + (long)(col0 + lr) * K + hk;

    v8f acc = {};
#pragma unroll 4
    for (int k0 = 0; k0 < K; k0 += 32) {
        const float4 f0 = *(const float4*)(arow + k0 + 0);
        const float4 f1 = *(const float4*)(arow + k0 + 4);
        const float4 f2 = *(const float4*)(arow + k0 + 8);
        const float4 f3 = *(const float4*)(arow + k0 + 12);
        const v8h b0 = *(const v8h*)(brow + k0);
        const v8h b1 = *(const v8h*)(brow + k0 + 8);

        v16h a;
        a[0]=(_Float16)f0.x;  a[1]=(_Float16)f0.y;  a[2]=(_Float16)f0.z;  a[3]=(_Float16)f0.w;
        a[4]=(_Float16)f1.x;  a[5]=(_Float16)f1.y;  a[6]=(_Float16)f1.z;  a[7]=(_Float16)f1.w;
        a[8]=(_Float16)f2.x;  a[9]=(_Float16)f2.y;  a[10]=(_Float16)f2.z; a[11]=(_Float16)f2.w;
        a[12]=(_Float16)f3.x; a[13]=(_Float16)f3.y; a[14]=(_Float16)f3.z; a[15]=(_Float16)f3.w;
        v16h b = __builtin_shufflevector(b0, b1, 0,1,2,3,4,5,6,7,8,9,10,11,12,13,14,15);

        acc = __builtin_amdgcn_wmma_f32_16x16x32_f16(
                  false, a, false, b, (short)0, acc, false, false);
    }

    const int orow = row0 + ((lane >> 4) << 3);   // lanes 16-31 hold M=8..15
    const int ocol = col0 + lr;
    const float bv = bias ? bias[ocol] : 0.f;
#pragma unroll
    for (int r = 0; r < 8; ++r) {
        float v = acc[r] + bv;
        if (relu) v = fmaxf(v, 0.f);
        C[(long)(orow + r) * Nc + ocol] = v;
    }
}

// ---------------------------------------------------------------- GCN scatter
// agg[dst] += hw[src] * dinv[src]*dinv[dst]; 32 lanes per edge, 4 chans/lane.
__global__ void gcn_scatter_kernel(const float* __restrict__ hw,
                                   const int* __restrict__ ei,
                                   const float* __restrict__ dinv,
                                   float* __restrict__ agg, int E)
{
    long tid  = (long)blockIdx.x * blockDim.x + threadIdx.x;
    long edge = tid >> 5;
    if (edge >= (long)E + NN_) return;
    int c = (int)(tid & 31) * 4;

    int s, d;
    if (edge < E) { s = ei[edge]; d = ei[E + edge]; }
    else          { s = d = (int)(edge - E); }
    float nrm = dinv[s] * dinv[d];

    const float4 hv = *(const float4*)(hw + (long)s * HID_ + c);
    float* ap = agg + (long)d * HID_ + c;
    atomicAdd(ap + 0, hv.x * nrm);
    atomicAdd(ap + 1, hv.y * nrm);
    atomicAdd(ap + 2, hv.z * nrm);
    atomicAdd(ap + 3, hv.w * nrm);
}

__global__ void bias_relu_kernel(float* __restrict__ h, const float* __restrict__ b, long n) {
    long i = (long)blockIdx.x * blockDim.x + threadIdx.x;
    if (i < n) { int c = (int)(i & (HID_ - 1)); h[i] = fmaxf(h[i] + b[c], 0.f); }
}

// ---------------------------------------------------------------- mean pool over stations
__global__ __launch_bounds__(HID_) void pool_kernel(const float* __restrict__ h,
                                                    float* __restrict__ out) {
    int bt = blockIdx.x, c = threadIdx.x;
    const float* p = h + (long)bt * S_ * HID_ + c;
    float s = 0.f;
    for (int st = 0; st < S_; ++st) s += p[(long)st * HID_];
    out[bt * HID_ + c] = s * (1.0f / S_);
}

// ---------------------------------------------------------------- attention
// one wave per (batch, head); lane = head-dim channel (HD_ == 32)
__global__ __launch_bounds__(32) void attention_kernel(const float* __restrict__ Q,
                                                       const float* __restrict__ K,
                                                       const float* __restrict__ V,
                                                       float* __restrict__ O) {
    int b  = blockIdx.x / NHEADS_;
    int hd = blockIdx.x % NHEADS_;
    int lane = threadIdx.x;
    const float scale = 0.17677669529663687f;  // 1/sqrt(32)

    float qv[WIN_], kv[WIN_], vv[WIN_];
#pragma unroll
    for (int t = 0; t < WIN_; ++t) {
        long idx = (long)(b * WIN_ + t) * HID_ + hd * HD_ + lane;
        qv[t] = Q[idx]; kv[t] = K[idx]; vv[t] = V[idx];
    }

    for (int t = 0; t < WIN_; ++t) {
        float sc[WIN_]; float mx = -3.4e38f;
#pragma unroll
        for (int u = 0; u < WIN_; ++u) {
            float p = qv[t] * kv[u];
#pragma unroll
            for (int off = 16; off >= 1; off >>= 1) p += __shfl_xor(p, off, 32);
            sc[u] = p * scale;
            mx = fmaxf(mx, sc[u]);
        }
        float den = 0.f;
#pragma unroll
        for (int u = 0; u < WIN_; ++u) { sc[u] = __expf(sc[u] - mx); den += sc[u]; }
        float inv = 1.f / den, o = 0.f;
#pragma unroll
        for (int u = 0; u < WIN_; ++u) o += sc[u] * inv * vv[u];
        O[(long)(b * WIN_ + t) * HID_ + hd * HD_ + lane] = o;
    }
}

// ---------------------------------------------------------------- residual + LayerNorm (in place on h)
__global__ __launch_bounds__(HID_) void add_ln_kernel(float* __restrict__ h,
                                                      const float* __restrict__ y,
                                                      const float* __restrict__ g,
                                                      const float* __restrict__ b) {
    int row = blockIdx.x, c = threadIdx.x;
    long idx = (long)row * HID_ + c;
    float x = h[idx] + y[idx];

    __shared__ float red[4];
    float s = x;
#pragma unroll
    for (int off = 16; off >= 1; off >>= 1) s += __shfl_xor(s, off, 32);
    if ((c & 31) == 0) red[c >> 5] = s;
    __syncthreads();
    float mean = (red[0] + red[1] + red[2] + red[3]) * (1.0f / HID_);
    __syncthreads();

    float d  = x - mean;
    float s2 = d * d;
#pragma unroll
    for (int off = 16; off >= 1; off >>= 1) s2 += __shfl_xor(s2, off, 32);
    if ((c & 31) == 0) red[c >> 5] = s2;
    __syncthreads();
    float var = (red[0] + red[1] + red[2] + red[3]) * (1.0f / HID_);

    h[idx] = d * rsqrtf(var + LN_EPS) * g[c] + b[c];
}

// ---------------------------------------------------------------- final FC: out[b] = h[b,last,:]·Wfc + bfc
__global__ __launch_bounds__(HID_) void final_fc_kernel(const float* __restrict__ h,
                                                        const float* __restrict__ W,
                                                        const float* __restrict__ bfc,
                                                        float* __restrict__ out) {
    int b = blockIdx.x, c = threadIdx.x;
    float p = h[(long)(b * WIN_ + WIN_ - 1) * HID_ + c] * W[c];
    __shared__ float red[4];
#pragma unroll
    for (int off = 16; off >= 1; off >>= 1) p += __shfl_xor(p, off, 32);
    if ((c & 31) == 0) red[c >> 5] = p;
    __syncthreads();
    if (c == 0) out[b] = red[0] + red[1] + red[2] + red[3] + bfc[0];
}

// ================================================================ launch
extern "C" void kernel_launch(void* const* d_in, const int* in_sizes, int n_in,
                              void* d_out, int out_size, void* d_ws, size_t ws_size,
                              hipStream_t stream) {
    const float* x  = (const float*)d_in[0];
    const int*   ei = (const int*)d_in[1];
    const int    E  = in_sizes[1] / 2;

    const float *gW[3], *gB[3];
    for (int l = 0; l < 3; ++l) { gW[l] = (const float*)d_in[3 + 2*l]; gB[l] = (const float*)d_in[4 + 2*l]; }
    const float *Wq[3], *bq[3], *Wk[3], *bk[3], *Wv[3], *bv[3], *Wo[3], *bo[3],
                *g1[3], *b1[3], *W1[3], *bf1[3], *W2[3], *bf2[3], *g2[3], *b2[3];
    for (int l = 0; l < 3; ++l) {
        int base = 9 + 16 * l;
        Wq[l]  = (const float*)d_in[base + 0];  bq[l]  = (const float*)d_in[base + 1];
        Wk[l]  = (const float*)d_in[base + 2];  bk[l]  = (const float*)d_in[base + 3];
        Wv[l]  = (const float*)d_in[base + 4];  bv[l]  = (const float*)d_in[base + 5];
        Wo[l]  = (const float*)d_in[base + 6];  bo[l]  = (const float*)d_in[base + 7];
        g1[l]  = (const float*)d_in[base + 8];  b1[l]  = (const float*)d_in[base + 9];
        W1[l]  = (const float*)d_in[base + 10]; bf1[l] = (const float*)d_in[base + 11];
        W2[l]  = (const float*)d_in[base + 12]; bf2[l] = (const float*)d_in[base + 13];
        g2[l]  = (const float*)d_in[base + 14]; b2[l]  = (const float*)d_in[base + 15];
    }
    const float* Wfc = (const float*)d_in[57];
    const float* bfc = (const float*)d_in[58];

    // ---- carve workspace
    char* ws = (char*)d_ws; size_t off = 0;
    auto carve = [&](size_t bytes) -> void* {
        void* p = (void*)(ws + off);
        off += (bytes + 255) & ~(size_t)255;
        return p;
    };
    float*     bufA = (float*)    carve((size_t)NN_ * HID_ * 4);   // hw (gather side)
    float*     bufB = (float*)    carve((size_t)NN_ * HID_ * 4);   // h / agg
    float*     xpad = (float*)    carve((size_t)NN_ * 32 * 4);     // padded input [N,32]
    float*     dinv = (float*)    carve((size_t)NN_ * 4);
    float*     hbt  = (float*)    carve((size_t)BT_ * HID_ * 4);
    float*     q    = (float*)    carve((size_t)BT_ * HID_ * 4);
    float*     k    = (float*)    carve((size_t)BT_ * HID_ * 4);
    float*     v    = (float*)    carve((size_t)BT_ * HID_ * 4);
    float*     attn = (float*)    carve((size_t)BT_ * HID_ * 4);
    float*     proj = (float*)    carve((size_t)BT_ * HID_ * 4);
    float*     ff   = (float*)    carve((size_t)BT_ * DFF_ * 4);
    _Float16*  Bt   = (_Float16*) carve((size_t)DFF_ * HID_ * 2);  // staged weight (max 512 KB)
    (void)ws_size; (void)n_in; (void)out_size;

    const long NH = (long)NN_ * HID_;

    auto stage = [&](const float* W, int K, int Nc, int Ksrc) {
        long n = (long)Nc * K;
        conv_bt_kernel<<<(int)((n + 255) / 256), 256, 0, stream>>>(W, Bt, K, Nc, Ksrc);
    };

    // ---- degree -> D^-1/2
    fill_zero_kernel<<<(NN_ + 255) / 256, 256, 0, stream>>>(dinv, NN_);
    deg_count_kernel<<<(E + NN_ + 255) / 256, 256, 0, stream>>>(ei, dinv, E);
    deg_to_dinv_kernel<<<(NN_ + 255) / 256, 256, 0, stream>>>(dinv);

    // ---- pad x to [N,32]
    pad_x_kernel<<<(int)(((long)NN_ * 32 + 255) / 256), 256, 0, stream>>>(x, xpad);

    // ---- GCN layers
    dim3 gg(NN_ / 16, HID_ / 64);
    for (int l = 0; l < 3; ++l) {
        if (l == 0) {
            stage(gW[0], 32, HID_, F_);
            wmma_gemm_kernel<32><<<gg, 128, 0, stream>>>(xpad, Bt, nullptr, bufA, HID_, 0);
        } else {
            stage(gW[l], HID_, HID_, HID_);
            wmma_gemm_kernel<HID_><<<gg, 128, 0, stream>>>(bufB, Bt, nullptr, bufA, HID_, 0);
        }
        fill_zero_kernel<<<(int)((NH + 255) / 256), 256, 0, stream>>>(bufB, NH);
        long sthreads = (long)(E + NN_) * 32;
        gcn_scatter_kernel<<<(int)((sthreads + 255) / 256), 256, 0, stream>>>(bufA, ei, dinv, bufB, E);
        bias_relu_kernel<<<(int)((NH + 255) / 256), 256, 0, stream>>>(bufB, gB[l], NH);
    }

    // ---- mean pool over stations -> [96, 128]
    pool_kernel<<<BT_, HID_, 0, stream>>>(bufB, hbt);

    // ---- transformer layers
    dim3 gs(BT_ / 16, HID_ / 64);
    dim3 gf(BT_ / 16, DFF_ / 64);
    for (int l = 0; l < 3; ++l) {
        stage(Wq[l], HID_, HID_, HID_);
        wmma_gemm_kernel<HID_><<<gs, 128, 0, stream>>>(hbt, Bt, bq[l], q, HID_, 0);
        stage(Wk[l], HID_, HID_, HID_);
        wmma_gemm_kernel<HID_><<<gs, 128, 0, stream>>>(hbt, Bt, bk[l], k, HID_, 0);
        stage(Wv[l], HID_, HID_, HID_);
        wmma_gemm_kernel<HID_><<<gs, 128, 0, stream>>>(hbt, Bt, bv[l], v, HID_, 0);
        attention_kernel<<<NB_ * NHEADS_, 32, 0, stream>>>(q, k, v, attn);
        stage(Wo[l], HID_, HID_, HID_);
        wmma_gemm_kernel<HID_><<<gs, 128, 0, stream>>>(attn, Bt, bo[l], proj, HID_, 0);
        add_ln_kernel<<<BT_, HID_, 0, stream>>>(hbt, proj, g1[l], b1[l]);
        stage(W1[l], HID_, DFF_, HID_);
        wmma_gemm_kernel<HID_><<<gf, 128, 0, stream>>>(hbt, Bt, bf1[l], ff, DFF_, 1);
        stage(W2[l], DFF_, HID_, DFF_);
        wmma_gemm_kernel<DFF_><<<gs, 128, 0, stream>>>(ff, Bt, bf2[l], proj, HID_, 0);
        add_ln_kernel<<<BT_, HID_, 0, stream>>>(hbt, proj, g2[l], b2[l]);
    }

    // ---- head
    final_fc_kernel<<<NB_, HID_, 0, stream>>>(hbt, Wfc, bfc, (float*)d_out);
}
